// GraphEdgePredictor_60730837565673
// MI455X (gfx1250) — matrix operand order repaired and checked by reading, not compiled, not run on hardware
//
#include <hip/hip_runtime.h>
#include <hip/hip_bf16.h>
#include <math.h>

// ---------------------------------------------------------------------------
// Problem constants (from reference)
// ---------------------------------------------------------------------------
constexpr int NB   = 8;     // batch
constexpr int NN   = 1024;  // nodes
constexpr int CIN  = 128;
constexpr int HH   = 256;   // hidden
constexpr int NHEADS = 8;
constexpr int HDIM = 32;    // head dim
constexpr int NLAYERS = 4;
constexpr int TOK  = NB * NN;           // 8192 tokens
constexpr long long EDGE_ELEMS = (long long)NB * NN * NN;  // 8388608

typedef __bf16 bf16_t;
typedef __attribute__((ext_vector_type(16))) __bf16 bf16x16;
typedef __attribute__((ext_vector_type(8)))  __bf16 bf16x8;
typedef __attribute__((ext_vector_type(8)))  float  f32x8;

union BFrag {
    bf16x8  h[2];
    bf16x16 v;
};

// ---------------------------------------------------------------------------
// Wave reductions (wave32)
// ---------------------------------------------------------------------------
__device__ inline float waveSum(float v) {
    for (int o = 16; o > 0; o >>= 1) v += __shfl_xor(v, o, 32);
    return v;
}
__device__ inline float waveMax(float v) {
    for (int o = 16; o > 0; o >>= 1) v = fmaxf(v, __shfl_xor(v, o, 32));
    return v;
}

// ---------------------------------------------------------------------------
// WMMA fragment helpers
// A fragment (16x32, bf16): lane<16 holds K[0..7]+K[16..23] of row l16,
// lane>=16 holds K[8..15]+K[24..31]  -> two contiguous b128 loads.
// B fragment (32x16 of W^T with W in (N,K) row-major): lane<16 holds K[0..7]+K[16..23]
// of col l16, lane>=16 holds K[8..15]+K[24..31] -> two contiguous b128 loads.
// ---------------------------------------------------------------------------
__device__ __forceinline__ void loadAFrag(BFrag& f, const bf16_t* p) {
    f.h[0] = *(const bf16x8*)(p);
    f.h[1] = *(const bf16x8*)(p + 16);
}
__device__ __forceinline__ void loadBFrag(BFrag& f, const bf16_t* p) {
    f.h[0] = *(const bf16x8*)(p);
    f.h[1] = *(const bf16x8*)(p + 16);
}

template<int NFR>
__device__ __forceinline__ void loadTile(BFrag (&a)[2], BFrag (&b)[NFR],
                                         const bf16_t* (&pA)[2], const bf16_t* (&pB)[NFR]) {
#pragma unroll
    for (int i = 0; i < 2; ++i) { loadAFrag(a[i], pA[i]); pA[i] += 32; }
#pragma unroll
    for (int j = 0; j < NFR; ++j) { loadBFrag(b[j], pB[j]); pB[j] += 32; }
}

template<int NFR>
__device__ __forceinline__ void computeTile(f32x8 (&acc)[2][NFR],
                                            const BFrag (&a)[2], const BFrag (&b)[NFR]) {
#pragma unroll
    for (int i = 0; i < 2; ++i)
#pragma unroll
        for (int j = 0; j < NFR; ++j)
            acc[i][j] = __builtin_amdgcn_wmma_f32_16x16x32_bf16(
                false, a[i].v, false, b[j].v, (short)0, acc[i][j], false, false);
}

// ---------------------------------------------------------------------------
// Generic batched bf16 WMMA GEMM:
//   C[z] = act( scale * A[z](MxK) @ W[z](NxK)^T + bias )
// Block = 4 waves (128 thr); wave computes 32(M) x NFR*16(N);
// block tile 128 x NFR*16.  M must be a multiple of 128, N of NFR*16,
// K of 32 (all true for this model).
// Batch offset: z -> (zo = z/zdiv, zi = z%zdiv), ptr += zo*outer + zi*inner.
// K-loop is branch-free and double-buffered (load next tile while 8 WMMAs
// run on the current one).
// ---------------------------------------------------------------------------
template<int NFR>
__global__ __launch_bounds__(128)
void gemm_bf16_wmma(const bf16_t* __restrict__ A, long long aOuter, long long aInner, int ldA,
                    const bf16_t* __restrict__ W, long long wOuter, long long wInner, int ldW,
                    float* __restrict__ C, long long cOuter, long long cInner, int ldC,
                    const float* __restrict__ bias,
                    int K, float scale, int doRelu, int zdiv)
{
    const int z  = blockIdx.z;
    const int zo = z / zdiv;
    const int zi = z - zo * zdiv;
    A += (long long)zo * aOuter + (long long)zi * aInner;
    W += (long long)zo * wOuter + (long long)zi * wInner;
    C += (long long)zo * cOuter + (long long)zi * cInner;

    const int wave = threadIdx.x >> 5;
    const int lane = threadIdx.x & 31;
    const int half = lane >> 4;       // 0 or 1
    const int l16  = lane & 15;

    const int mWave = blockIdx.y * 128 + wave * 32;   // 32 rows per wave
    const int nWave = blockIdx.x * (NFR * 16);

    const bf16_t* pA[2];
#pragma unroll
    for (int i = 0; i < 2; ++i)
        pA[i] = A + (long long)(mWave + i * 16 + l16) * ldA + half * 8;
    const bf16_t* pB[NFR];
#pragma unroll
    for (int j = 0; j < NFR; ++j)
        pB[j] = W + (long long)(nWave + j * 16 + l16) * ldW + half * 8;

    f32x8 acc[2][NFR];
#pragma unroll
    for (int i = 0; i < 2; ++i)
#pragma unroll
        for (int j = 0; j < NFR; ++j)
#pragma unroll
            for (int e = 0; e < 8; ++e) acc[i][j][e] = 0.0f;

    BFrag a0[2], b0[NFR], a1[2], b1[NFR];

    loadTile<NFR>(a0, b0, pA, pB);
    int s = (K >> 5) - 1;
    while (s >= 2) {
        loadTile<NFR>(a1, b1, pA, pB);
        computeTile<NFR>(acc, a0, b0);
        loadTile<NFR>(a0, b0, pA, pB);
        computeTile<NFR>(acc, a1, b1);
        s -= 2;
    }
    if (s == 1) {
        loadTile<NFR>(a1, b1, pA, pB);
        computeTile<NFR>(acc, a0, b0);
        computeTile<NFR>(acc, a1, b1);
    } else {
        computeTile<NFR>(acc, a0, b0);
    }

    // Store: lane<16 -> rows mBase+0..7, lane>=16 -> rows mBase+8..15
#pragma unroll
    for (int i = 0; i < 2; ++i) {
#pragma unroll
        for (int j = 0; j < NFR; ++j) {
            const int n = nWave + j * 16 + l16;
            const int mBase = mWave + i * 16 + half * 8;
            const float bv = bias ? bias[n] : 0.0f;
#pragma unroll
            for (int r = 0; r < 8; ++r) {
                float v = acc[i][j][r] * scale + bv;
                if (doRelu) v = fmaxf(v, 0.0f);
                C[(long long)(mBase + r) * ldC + n] = v;
            }
        }
    }
}

// ---------------------------------------------------------------------------
// Elementwise helpers
// ---------------------------------------------------------------------------
__global__ void k_convert_f32_bf16(const float* __restrict__ in, bf16_t* __restrict__ out,
                                   long long n) {
    for (long long i = blockIdx.x * (long long)blockDim.x + threadIdx.x; i < n;
         i += (long long)gridDim.x * blockDim.x)
        out[i] = (bf16_t)in[i];
}

__global__ void k_adjsum_bf16(const float* __restrict__ a, const float* __restrict__ b,
                              bf16_t* __restrict__ out, long long n) {
    for (long long i = blockIdx.x * (long long)blockDim.x + threadIdx.x; i < n;
         i += (long long)gridDim.x * blockDim.x)
        out[i] = (bf16_t)(0.5f * (a[i] + b[i]));
}

// Per-batch transpose fp32(rows x cols) -> bf16(cols x rows), LDS-tiled.
__global__ void k_transpose_f32_bf16(const float* __restrict__ in, bf16_t* __restrict__ out,
                                     int rows, int cols) {
    __shared__ float tile[32][33];
    const long long bofs = (long long)blockIdx.z * rows * cols;
    in  += bofs;
    out += bofs;
    const int bR = blockIdx.y * 32, bC = blockIdx.x * 32;
    const int tx = threadIdx.x, ty = threadIdx.y;   // 32 x 8
    for (int i = ty; i < 32; i += 8) {
        int r = bR + i, c = bC + tx;
        tile[i][tx] = (r < rows && c < cols) ? in[(long long)r * cols + c] : 0.0f;
    }
    __syncthreads();
    for (int i = ty; i < 32; i += 8) {
        int oc = bC + i, orow = bR + tx;  // out[col][row]
        if (oc < cols && orow < rows)
            out[(long long)oc * rows + orow] = (bf16_t)tile[tx][i];
    }
}

// qkv (TOK x 768) fp32 -> qh/kh (B,NH,N,HD) bf16 and vT (B,NH,HD,N) bf16
__global__ void k_split_qkv(const float* __restrict__ qkv, bf16_t* __restrict__ qh,
                            bf16_t* __restrict__ kh, bf16_t* __restrict__ vT) {
    long long idx = blockIdx.x * (long long)blockDim.x + threadIdx.x;
    const long long total = (long long)TOK * HH;
    if (idx >= total) return;
    const int c   = (int)(idx & (HH - 1));
    const long long tkn = idx >> 8;           // /256
    const int n = (int)(tkn & (NN - 1));
    const int b = (int)(tkn >> 10);
    const int h = c >> 5, d = c & 31;
    const float* base = qkv + tkn * (3 * HH);
    const long long bh = (long long)b * NHEADS + h;
    qh[(bh * NN + n) * HDIM + d] = (bf16_t)base[c];
    kh[(bh * NN + n) * HDIM + d] = (bf16_t)base[HH + c];
    vT[(bh * HDIM + d) * NN + n] = (bf16_t)base[2 * HH + c];
}

// Row softmax with key-padding mask; s (64,1024,1024) fp32 -> probs bf16.
__global__ __launch_bounds__(256)
void k_softmax_mask(const float* __restrict__ s, bf16_t* __restrict__ p,
                    const int* __restrict__ node_mask) {
    const long long row = blockIdx.x;          // 64*1024 rows
    const int b = (int)(row >> 13);            // row / (NHEADS*NN)
    const float* srow = s + row * NN;
    bf16_t* prow = p + row * NN;
    const int* mrow = node_mask + b * NN;
    __shared__ float redMax[8], redSum[8];
    const int t = threadIdx.x;
    const float NEG = -__builtin_inff();

    float vals[4];
    float mx = NEG;
#pragma unroll
    for (int i = 0; i < 4; ++i) {
        int j = t + i * 256;
        float v = (mrow[j] == 0) ? NEG : srow[j];
        vals[i] = v;
        mx = fmaxf(mx, v);
    }
    mx = waveMax(mx);
    if ((t & 31) == 0) redMax[t >> 5] = mx;
    __syncthreads();
    mx = redMax[0];
#pragma unroll
    for (int w = 1; w < 8; ++w) mx = fmaxf(mx, redMax[w]);

    float ex[4];
    float sum = 0.0f;
#pragma unroll
    for (int i = 0; i < 4; ++i) {
        float e = (vals[i] == NEG) ? 0.0f : __expf(vals[i] - mx);
        ex[i] = e;
        sum += e;
    }
    sum = waveSum(sum);
    if ((t & 31) == 0) redSum[t >> 5] = sum;
    __syncthreads();
    sum = 0.0f;
#pragma unroll
    for (int w = 0; w < 8; ++w) sum += redSum[w];
    const float inv = (sum > 0.0f) ? (1.0f / sum) : 0.0f;
#pragma unroll
    for (int i = 0; i < 4; ++i)
        prow[t + i * 256] = (bf16_t)(ex[i] * inv);
}

// z = LayerNorm(z + t) * g + b ; one wave per row of H=256 (8 vals/lane)
__global__ __launch_bounds__(32)
void k_residual_ln(float* __restrict__ z, const float* __restrict__ t,
                   const float* __restrict__ g, const float* __restrict__ b) {
    const long long row = blockIdx.x;
    float* zr = z + row * HH;
    const float* tr = t + row * HH;
    const int lane = threadIdx.x;
    float v[8];
    float sum = 0.0f;
#pragma unroll
    for (int i = 0; i < 8; ++i) {
        v[i] = zr[lane + i * 32] + tr[lane + i * 32];
        sum += v[i];
    }
    const float mu = waveSum(sum) * (1.0f / HH);
    float vs = 0.0f;
#pragma unroll
    for (int i = 0; i < 8; ++i) {
        float d = v[i] - mu;
        vs += d * d;
    }
    const float var = waveSum(vs) * (1.0f / HH);
    const float rs = rsqrtf(var + 1e-5f);
#pragma unroll
    for (int i = 0; i < 8; ++i) {
        int c = lane + i * 32;
        zr[c] = (v[i] - mu) * rs * g[c] + b[c];
    }
}

// za[t] = z[t,:]·w1 ; zb[t] = z[t,:]·w2   (w1 = dec_w[0:256], w2 = dec_w[256:512])
__global__ __launch_bounds__(32)
void k_dec_proj(const float* __restrict__ z, const float* __restrict__ dec_w,
                float* __restrict__ za, float* __restrict__ zb) {
    const long long row = blockIdx.x;
    const float* zr = z + row * HH;
    const int lane = threadIdx.x;
    float s1 = 0.0f, s2 = 0.0f;
#pragma unroll
    for (int i = 0; i < 8; ++i) {
        int c = lane + i * 32;
        float zv = zr[c];
        s1 += zv * dec_w[c];
        s2 += zv * dec_w[HH + c];
    }
    s1 = waveSum(s1);
    s2 = waveSum(s2);
    if (lane == 0) { za[row] = s1; zb[row] = s2; }
}

// edge_logits[b,i,j] = za[b,i] + zb[b,j] + dec_b
__global__ void k_edge(const float* __restrict__ za, const float* __restrict__ zb,
                       const float* __restrict__ dec_b, float* __restrict__ out) {
    const float bias = dec_b[0];
    for (long long idx = blockIdx.x * (long long)blockDim.x + threadIdx.x; idx < EDGE_ELEMS;
         idx += (long long)gridDim.x * blockDim.x) {
        const int b = (int)(idx >> 20);
        const int i = (int)((idx >> 10) & (NN - 1));
        const int j = (int)(idx & (NN - 1));
        out[idx] = za[b * NN + i] + zb[b * NN + j] + bias;
    }
}

__global__ void k_copy_f32(const float* __restrict__ in, float* __restrict__ out, long long n) {
    for (long long i = blockIdx.x * (long long)blockDim.x + threadIdx.x; i < n;
         i += (long long)gridDim.x * blockDim.x)
        out[i] = in[i];
}

// ---------------------------------------------------------------------------
// Host-side orchestration
// ---------------------------------------------------------------------------
template<int NFR>
static inline dim3 gemmGrid(int M, int N, int batch) {
    return dim3(N / (NFR * 16), M / 128, batch);
}

extern "C" void kernel_launch(void* const* d_in, const int* in_sizes, int n_in,
                              void* d_out, int out_size, void* d_ws, size_t ws_size,
                              hipStream_t stream) {
    (void)in_sizes; (void)n_in; (void)out_size; (void)ws_size;

    const float* x        = (const float*)d_in[0];
    const float* adj_out  = (const float*)d_in[1];
    const float* adj_in   = (const float*)d_in[2];
    const int*   node_mask= (const int*)  d_in[3];
    const float* fc1_w    = (const float*)d_in[4];
    const float* fc1_b    = (const float*)d_in[5];
    const float* fc2_w    = (const float*)d_in[6];
    const float* fc2_b    = (const float*)d_in[7];
    const float* attn_in_w  = (const float*)d_in[8];
    const float* attn_in_b  = (const float*)d_in[9];
    const float* attn_out_w = (const float*)d_in[10];
    const float* attn_out_b = (const float*)d_in[11];
    const float* lin_w    = (const float*)d_in[12];
    const float* lin_b    = (const float*)d_in[13];
    const float* ln1_w    = (const float*)d_in[14];
    const float* ln1_b    = (const float*)d_in[15];
    const float* ln2_w    = (const float*)d_in[16];
    const float* ln2_b    = (const float*)d_in[17];
    const float* dec_w    = (const float*)d_in[18];
    const float* dec_b    = (const float*)d_in[19];

    float* edge_out = (float*)d_out;                       // 8*1024*1024
    float* z_out    = (float*)d_out + EDGE_ELEMS;          // 8*1024*256

    // ---- workspace bump allocator (256B aligned) ----
    char* wp = (char*)d_ws;
    auto alloc = [&](size_t bytes) -> void* {
        void* p = wp;
        wp += (bytes + 255) & ~(size_t)255;
        return p;
    };
    bf16_t* x_bf    = (bf16_t*)alloc((size_t)TOK * CIN * 2);
    bf16_t* fc1w_bf = (bf16_t*)alloc((size_t)HH * CIN * 2);
    bf16_t* fc2w_bf = (bf16_t*)alloc((size_t)HH * HH * 2);
    bf16_t* aiw_bf  = (bf16_t*)alloc((size_t)NLAYERS * 3 * HH * HH * 2);
    bf16_t* aow_bf  = (bf16_t*)alloc((size_t)NLAYERS * HH * HH * 2);
    bf16_t* lw_bf   = (bf16_t*)alloc((size_t)NLAYERS * HH * HH * 2);
    float*  hbuf    = (float*) alloc((size_t)TOK * HH * 4);
    bf16_t* adjsum  = (bf16_t*)alloc((size_t)NB * NN * NN * 2);
    bf16_t* hT_bf   = (bf16_t*)alloc((size_t)NB * HH * NN * 2);
    float*  h2buf   = (float*) alloc((size_t)TOK * HH * 4);
    float*  zbuf    = (float*) alloc((size_t)TOK * HH * 4);
    bf16_t* zb_bf   = (bf16_t*)alloc((size_t)TOK * HH * 2);
    float*  qkvbuf  = (float*) alloc((size_t)TOK * 3 * HH * 4);
    bf16_t* qh_bf   = (bf16_t*)alloc((size_t)NB * NHEADS * NN * HDIM * 2);
    bf16_t* kh_bf   = (bf16_t*)alloc((size_t)NB * NHEADS * NN * HDIM * 2);
    bf16_t* vT_bf   = (bf16_t*)alloc((size_t)NB * NHEADS * HDIM * NN * 2);
    float*  sbuf    = (float*) alloc((size_t)NB * NHEADS * NN * NN * 4);   // 256 MB
    bf16_t* abf     = (bf16_t*)alloc((size_t)NB * NHEADS * NN * NN * 2);   // 128 MB
    float*  obuf    = (float*) alloc((size_t)TOK * HH * 4);
    bf16_t* ob_bf   = (bf16_t*)alloc((size_t)TOK * HH * 2);
    float*  tmpbuf  = (float*) alloc((size_t)TOK * HH * 4);
    float*  za      = (float*) alloc((size_t)TOK * 4);
    float*  zbv     = (float*) alloc((size_t)TOK * 4);

    const dim3 cvtGrid(4096), cvtBlk(256);
    auto convert = [&](const float* in, bf16_t* out, long long n) {
        k_convert_f32_bf16<<<cvtGrid, cvtBlk, 0, stream>>>(in, out, n);
    };

    // ---- stage 0: precision conversion of inputs/weights ----
    convert(x, x_bf, (long long)TOK * CIN);
    convert(fc1_w, fc1w_bf, (long long)HH * CIN);
    convert(fc2_w, fc2w_bf, (long long)HH * HH);
    convert(attn_in_w, aiw_bf, (long long)NLAYERS * 3 * HH * HH);
    convert(attn_out_w, aow_bf, (long long)NLAYERS * HH * HH);
    convert(lin_w, lw_bf, (long long)NLAYERS * HH * HH);
    k_adjsum_bf16<<<cvtGrid, cvtBlk, 0, stream>>>(adj_out, adj_in, adjsum,
                                                  (long long)NB * NN * NN);

    // ---- fc1: h = relu(x @ fc1_w^T + fc1_b)  [8192 x 128 -> 256] ----
    gemm_bf16_wmma<4><<<gemmGrid<4>(TOK, HH, 1), 128, 0, stream>>>(
        x_bf, 0, 0, CIN, fc1w_bf, 0, 0, CIN, hbuf, 0, 0, HH,
        fc1_b, CIN, 1.0f, 1, 1);

    // ---- adjacency aggregation: h2 = adjsum @ h (per batch) ----
    k_transpose_f32_bf16<<<dim3(HH / 32, NN / 32, NB), dim3(32, 8), 0, stream>>>(
        hbuf, hT_bf, NN, HH);
    gemm_bf16_wmma<4><<<gemmGrid<4>(NN, HH, NB), 128, 0, stream>>>(
        adjsum, (long long)NN * NN, 0, NN,
        hT_bf, (long long)HH * NN, 0, NN,
        h2buf, (long long)NN * HH, 0, HH,
        nullptr, NN, 1.0f, 0, 1);

    // ---- fc2: z = relu(h2 @ fc2_w^T + fc2_b) ----
    convert(h2buf, zb_bf, (long long)TOK * HH);
    gemm_bf16_wmma<4><<<gemmGrid<4>(TOK, HH, 1), 128, 0, stream>>>(
        zb_bf, 0, 0, HH, fc2w_bf, 0, 0, HH, zbuf, 0, 0, HH,
        fc2_b, HH, 1.0f, 1, 1);

    const float attnScale = 1.0f / sqrtf((float)HDIM);
    const int NBH = NB * NHEADS;

    for (int l = 0; l < NLAYERS; ++l) {
        // qkv = z @ Wi^T + bi    [8192 x 256 -> 768]
        convert(zbuf, zb_bf, (long long)TOK * HH);
        gemm_bf16_wmma<4><<<gemmGrid<4>(TOK, 3 * HH, 1), 128, 0, stream>>>(
            zb_bf, 0, 0, HH,
            aiw_bf + (size_t)l * 3 * HH * HH, 0, 0, HH,
            qkvbuf, 0, 0, 3 * HH,
            attn_in_b + l * 3 * HH, HH, 1.0f, 0, 1);

        // split to per-head bf16 operands
        {
            long long total = (long long)TOK * HH;
            k_split_qkv<<<dim3((unsigned)((total + 255) / 256)), 256, 0, stream>>>(
                qkvbuf, qh_bf, kh_bf, vT_bf);
        }

        // s = scale * q @ k^T  (per b,h) [1024 x 32 -> 1024]
        gemm_bf16_wmma<4><<<gemmGrid<4>(NN, NN, NBH), 128, 0, stream>>>(
            qh_bf, (long long)NN * HDIM, 0, HDIM,
            kh_bf, (long long)NN * HDIM, 0, HDIM,
            sbuf, (long long)NN * NN, 0, NN,
            nullptr, HDIM, attnScale, 0, 1);

        // mask + softmax -> bf16 probs
        k_softmax_mask<<<dim3((unsigned)(NBH * NN)), 256, 0, stream>>>(sbuf, abf, node_mask);

        // o = a @ v  (per b,h) [1024 x 1024 -> 32], scattered into (B,N,H)
        gemm_bf16_wmma<2><<<gemmGrid<2>(NN, HDIM, NBH), 128, 0, stream>>>(
            abf, (long long)NHEADS * NN * NN, (long long)NN * NN, NN,
            vT_bf, (long long)NHEADS * HDIM * NN, (long long)HDIM * NN, NN,
            obuf, (long long)NN * HH, HDIM, HH,
            nullptr, NN, 1.0f, 0, NHEADS);

        // o @ Wo^T + bo -> tmp ; z = LN(z + tmp)
        convert(obuf, ob_bf, (long long)TOK * HH);
        gemm_bf16_wmma<4><<<gemmGrid<4>(TOK, HH, 1), 128, 0, stream>>>(
            ob_bf, 0, 0, HH,
            aow_bf + (size_t)l * HH * HH, 0, 0, HH,
            tmpbuf, 0, 0, HH,
            attn_out_b + l * HH, HH, 1.0f, 0, 1);
        k_residual_ln<<<dim3(TOK), 32, 0, stream>>>(zbuf, tmpbuf,
                                                    ln1_w + l * HH, ln1_b + l * HH);

        // z @ Wl^T + bl -> tmp ; z = LN(z + tmp)
        convert(zbuf, zb_bf, (long long)TOK * HH);
        gemm_bf16_wmma<4><<<gemmGrid<4>(TOK, HH, 1), 128, 0, stream>>>(
            zb_bf, 0, 0, HH,
            lw_bf + (size_t)l * HH * HH, 0, 0, HH,
            tmpbuf, 0, 0, HH,
            lin_b + l * HH, HH, 1.0f, 0, 1);
        k_residual_ln<<<dim3(TOK), 32, 0, stream>>>(zbuf, tmpbuf,
                                                    ln2_w + l * HH, ln2_b + l * HH);
    }

    // ---- decoder: rank-1 edge logits ----
    k_dec_proj<<<dim3(TOK), 32, 0, stream>>>(zbuf, dec_w, za, zbv);
    k_edge<<<dim3(16384), 256, 0, stream>>>(za, zbv, dec_b, edge_out);

    // ---- second output: z ----
    k_copy_f32<<<cvtGrid, cvtBlk, 0, stream>>>(zbuf, z_out, (long long)TOK * HH);
}
